// GraphEncoder_39539468927631
// MI455X (gfx1250) — matrix operand order repaired and checked by reading, not compiled, not run on hardware
//
#include <hip/hip_runtime.h>

// ---------------------------------------------------------------------------
// CDNA5 (gfx1250, wave32) implementation of the RNA graph encoder.
// Big GEMMs: v_wmma_f32_16x16x32_bf16 (f32 accumulate).
// GRU A-tile staging: global_load_async_to_lds_b128 (ASYNCcnt path).
// ---------------------------------------------------------------------------

typedef unsigned short u16;
typedef __bf16 bf16;
typedef bf16  v16bf __attribute__((ext_vector_type(16)));
typedef bf16  v8bf  __attribute__((ext_vector_type(8)));
typedef float v8f   __attribute__((ext_vector_type(8)));
typedef u16   v8u   __attribute__((ext_vector_type(8)));

#define M_MSG 196608
#define N_NUC 65536
#define HDIM  128

// ---- bf16 <-> f32 helpers (round-to-nearest-even) -------------------------
__device__ __forceinline__ u16 f2bf(float f) {
  union { float f; unsigned u; } x; x.f = f;
  unsigned r = x.u + 0x7FFFu + ((x.u >> 16) & 1u);
  return (u16)(r >> 16);
}
__device__ __forceinline__ float bf2f(u16 b) {
  union { unsigned u; float f; } x; x.u = ((unsigned)b) << 16;
  return x.f;
}
__device__ __forceinline__ float sigf(float x) { return 1.0f / (1.0f + expf(-x)); }

__device__ __forceinline__ v8f vzero8() {
  v8f z;
#pragma unroll
  for (int i = 0; i < 8; ++i) z[i] = 0.0f;
  return z;
}
__device__ __forceinline__ v16bf cat8(v8bf a, v8bf b) {
  return __builtin_shufflevector(a, b, 0,1,2,3,4,5,6,7,8,9,10,11,12,13,14,15);
}

// A-fragment (16x32 bf16) per ISA layout: lane l holds row (l&15);
// lanes 0-15: K = 0..7 (v0-3) and 16..23 (v4-7); lanes 16-31: +8.
__device__ __forceinline__ v16bf load_a_lds(const u16* lds, int row, int kbase, int ld) {
  const v8bf* p0 = reinterpret_cast<const v8bf*>(lds + row * ld + kbase);
  const v8bf* p1 = reinterpret_cast<const v8bf*>(lds + row * ld + kbase + 16);
  return cat8(*p0, *p1);
}
__device__ __forceinline__ v16bf load_a_glb(const u16* g, int row, int kbase, int ld) {
  const v8bf* p0 = reinterpret_cast<const v8bf*>(g + row * ld + kbase);
  const v8bf* p1 = reinterpret_cast<const v8bf*>(g + row * ld + kbase + 16);
  return cat8(*p0, *p1);
}
// B fragment from pre-packed weights: lane's 16 bf16 contiguous (32 B).
__device__ __forceinline__ v16bf load_b_pk(const u16* pk, int tile, int kStride, int ks, int lane) {
  const v16bf* p = reinterpret_cast<const v16bf*>(pk + (((tile * kStride + ks) * 32 + lane) << 4));
  return *p;
}
__device__ __forceinline__ v8f wmma_bf(v16bf a, v16bf b, v8f c) {
  return __builtin_amdgcn_wmma_f32_16x16x32_bf16(false, a, false, b, (short)0, c, false, false);
}

// Gather-sum of three bf16 rows, 8 elements at a time (16B loads).
__device__ __forceinline__ v8u gather3_8(const u16* msg, const int* g, int kc) {
  v8u a = *reinterpret_cast<const v8u*>(msg + g[0] * HDIM + kc);
  v8u b = *reinterpret_cast<const v8u*>(msg + g[1] * HDIM + kc);
  v8u d = *reinterpret_cast<const v8u*>(msg + g[2] * HDIM + kc);
  v8u o;
#pragma unroll
  for (int i = 0; i < 8; ++i) o[i] = f2bf(bf2f(a[i]) + bf2f(b[i]) + bf2f(d[i]));
  return o;
}

// ---------------------------------------------------------------------------
// Pack f32 weight matrix W [nOut x ldw] (row-major) into WMMA B-fragment
// layout: B[k][n] = W[n][k].
// ---------------------------------------------------------------------------
__global__ __launch_bounds__(256) void k_pack(u16* dst, int dstKStride, int ksBase,
                                              const float* W, int ldw, int n0,
                                              int kSteps, int k0, int total) {
  int idx = blockIdx.x * 256 + threadIdx.x;
  if (idx >= total) return;
  int per  = kSteps * 512;
  int ct   = idx / per;
  int rem  = idx - ct * per;
  int ks   = rem >> 9;
  int r2   = rem & 511;
  int lane = r2 >> 4;
  int ve   = r2 & 15;
  int n = n0 + ct * 16 + (lane & 15);
  int k = k0 + ks * 32 + (lane >> 4) * 8 + (ve & 7) + (ve >> 3) * 16;
  dst[((ct * dstKStride + ksBase + ks) * 32 + lane) * 16 + ve] = f2bf(W[n * ldw + k]);
}

// ---------------------------------------------------------------------------
// local = f_bond @ W_local.T  (K=8, scalar);  messages = relu(local)
// ---------------------------------------------------------------------------
__global__ __launch_bounds__(256) void k_local(const float* f_bond, const float* W_local,
                                               u16* local_bf, u16* msg_bf) {
  int e = blockIdx.x * 256 + threadIdx.x;
  if (e >= M_MSG * HDIM) return;
  int m = e >> 7, n = e & 127;
  const float* fb = f_bond + m * 8;
  const float* wl = W_local + n * 8;
  float v = 0.0f;
#pragma unroll
  for (int j = 0; j < 8; ++j) v = fmaf(fb[j], wl[j], v);
  local_bf[e] = f2bf(v);
  msg_bf[e]   = f2bf(fmaxf(v, 0.0f));
}

// ---------------------------------------------------------------------------
// GEMM 1 (fused neighbor gather): new_msg = relu(local + nei @ W_msg.T)
// Block: 64 rows x 128 cols; 8 waves, wave = 16 rows x 64 cols; K = 128.
// ---------------------------------------------------------------------------
#define ALD1 136   // padded LDS row stride (u16) for K=128
__global__ __launch_bounds__(256) void k_gemm_msg(const u16* __restrict__ msg,
                                                  const int* __restrict__ mgraph,
                                                  const u16* __restrict__ Bmsg,
                                                  const u16* __restrict__ local_bf,
                                                  u16* __restrict__ newmsg) {
  __shared__ u16 As[64 * ALD1];
  const int tid = threadIdx.x;
  const int rowBase = blockIdx.x * 64;
  // Stage A tile: nei[r][k] = sum of 3 gathered message rows (16B vector chunks)
  for (int c = tid; c < 64 * 16; c += 256) {
    int r = c >> 4, kc = (c & 15) * 8;
    const int* g = mgraph + (rowBase + r) * 3;
    *reinterpret_cast<v8u*>(As + r * ALD1 + kc) = gather3_8(msg, g, kc);
  }
  __syncthreads();

  const int lane = tid & 31, w = tid >> 5;
  const int rt = w >> 1;          // row tile 0..3
  const int ch = w & 1;           // col half 0..1
  const int row = rt * 16 + (lane & 15);
  const int khalf = (lane >> 4) * 8;
  v8f acc[4];
#pragma unroll
  for (int j = 0; j < 4; ++j) acc[j] = vzero8();

#pragma unroll
  for (int ks = 0; ks < 4; ++ks) {
    v16bf a = load_a_lds(As, row, ks * 32 + khalf, ALD1);
#pragma unroll
    for (int j = 0; j < 4; ++j) {
      v16bf b = load_b_pk(Bmsg, ch * 4 + j, 4, ks, lane);
      acc[j] = wmma_bf(a, b, acc[j]);
    }
  }
  // Epilogue: relu(local + acc) -> bf16
  const int hi = lane >> 4, nl = lane & 15;
#pragma unroll
  for (int j = 0; j < 4; ++j) {
    int n = (ch * 4 + j) * 16 + nl;
#pragma unroll
    for (int v = 0; v < 8; ++v) {
      int m = rowBase + rt * 16 + hi * 8 + v;
      float val = acc[j][v] + bf2f(local_bf[m * HDIM + n]);
      newmsg[m * HDIM + n] = f2bf(fmaxf(val, 0.0f));
    }
  }
}

// ---------------------------------------------------------------------------
// GEMM 2 + GRU (fused).  A = [new_msg | h] (K=256).  512 output columns:
//   cols   0..255 : r,z gates  = x@Wih_rz.T + h@Whh_rz.T   (ks 0..7)
//   cols 256..383 : i_n        = x@Wih_n.T                  (ks 0..3)
//   cols 384..511 : h_n        = h@Whh_n.T                  (ks 4..7)
// A tile staged via global_load_async_to_lds_b128 (ASYNCcnt, no VGPR round-trip).
// ---------------------------------------------------------------------------
#define ALD2 264   // padded LDS row stride (u16) for K=256
#define CLD  516   // padded LDS C row stride (f32)
__global__ __launch_bounds__(256) void k_gemm_gru(const u16* __restrict__ newmsg,
                                                  u16* __restrict__ msg,
                                                  const u16* __restrict__ Bgru,
                                                  const float* __restrict__ bih,
                                                  const float* __restrict__ bhh) {
  __shared__ u16  As[32 * ALD2];
  __shared__ float Cs[32 * CLD];
  const int tid = threadIdx.x;
  const int rowBase = blockIdx.x * 32;

  // Async copy: As[r][0..127] = new_msg row, As[r][128..255] = h row.
  // 16-byte chunks; LDS byte address carried in the "vdst" VGPR.
  for (int c = tid; c < 32 * 32; c += 256) {
    int r = c >> 5, k = (c & 31) * 8;
    int gr = rowBase + r;
    const u16* src = (k < 128) ? (newmsg + gr * HDIM + k)
                               : (msg + gr * HDIM + (k - 128));
    unsigned ldsb = (unsigned)(size_t)(As + r * ALD2 + k);
    asm volatile("global_load_async_to_lds_b128 %0, %1, off"
                 :: "v"(ldsb), "v"(src) : "memory");
  }
  asm volatile("s_wait_asynccnt 0x0" ::: "memory");
  __syncthreads();

  const int lane = tid & 31, w = tid >> 5;
  const int rt = w & 1;           // row tile
  const int cg = w >> 1;          // col group 0..3
  const int ks0 = (cg == 3) ? 4 : 0;
  const int ksN = (cg < 2) ? 8 : ks0 + 4;
  const int row = rt * 16 + (lane & 15);
  const int khalf = (lane >> 4) * 8;
  v8f acc[8];
#pragma unroll
  for (int j = 0; j < 8; ++j) acc[j] = vzero8();

  for (int ks = ks0; ks < ksN; ++ks) {
    v16bf a = load_a_lds(As, row, ks * 32 + khalf, ALD2);
#pragma unroll
    for (int j = 0; j < 8; ++j) {
      v16bf b = load_b_pk(Bgru, cg * 8 + j, 8, ks, lane);
      acc[j] = wmma_bf(a, b, acc[j]);
    }
  }
  // Stash C tile in LDS for cross-wave gate math
  const int hi = lane >> 4, nl = lane & 15;
#pragma unroll
  for (int j = 0; j < 8; ++j) {
#pragma unroll
    for (int v = 0; v < 8; ++v) {
      int m = rt * 16 + hi * 8 + v;
      int n = cg * 128 + j * 16 + nl;
      Cs[m * CLD + n] = acc[j][v];
    }
  }
  __syncthreads();

  // GRU elementwise + mask row 0; in-place bf16 state update, paired stores.
  for (int e = tid; e < 32 * 64; e += 256) {
    int r = e >> 6, jp = (e & 63) * 2;
    int gr = rowBase + r;
    const float* c = Cs + r * CLD;
    unsigned hpair = *reinterpret_cast<const unsigned*>(msg + gr * HDIM + jp);
    unsigned pk = 0;
#pragma unroll
    for (int q = 0; q < 2; ++q) {
      int j = jp + q;
      float rr = sigf(c[j]       + bih[j]       + bhh[j]);
      float zz = sigf(c[128 + j] + bih[128 + j] + bhh[128 + j]);
      float nn = tanhf(c[256 + j] + bih[256 + j] + rr * (c[384 + j] + bhh[256 + j]));
      float hp = bf2f((u16)(q ? (hpair >> 16) : (hpair & 0xffffu)));
      float hn = (1.0f - zz) * nn + zz * hp;
      if (gr == 0) hn = 0.0f;
      pk |= ((unsigned)f2bf(hn)) << (16 * q);
    }
    *reinterpret_cast<unsigned*>(msg + gr * HDIM + jp) = pk;
  }
}

// ---------------------------------------------------------------------------
// Node embedding: nuc_emb = relu([f_nuc, gather_sum(msg)] @ W_node.T)
// ---------------------------------------------------------------------------
__global__ __launch_bounds__(256) void k_gemm_node(const u16* __restrict__ msg,
                                                   const int* __restrict__ ngraph,
                                                   const u16* __restrict__ Bnode,
                                                   const float* __restrict__ f_nuc,
                                                   const float* __restrict__ W_node,
                                                   u16* __restrict__ nuc) {
  __shared__ u16 As[64 * ALD1];
  const int tid = threadIdx.x;
  const int rowBase = blockIdx.x * 64;
  for (int c = tid; c < 64 * 16; c += 256) {
    int r = c >> 4, kc = (c & 15) * 8;
    const int* g = ngraph + (rowBase + r) * 3;
    *reinterpret_cast<v8u*>(As + r * ALD1 + kc) = gather3_8(msg, g, kc);
  }
  __syncthreads();

  const int lane = tid & 31, w = tid >> 5;
  const int rt = w >> 1, ch = w & 1;
  const int row = rt * 16 + (lane & 15);
  const int khalf = (lane >> 4) * 8;
  v8f acc[4];
#pragma unroll
  for (int j = 0; j < 4; ++j) acc[j] = vzero8();
#pragma unroll
  for (int ks = 0; ks < 4; ++ks) {
    v16bf a = load_a_lds(As, row, ks * 32 + khalf, ALD1);
#pragma unroll
    for (int j = 0; j < 4; ++j) {
      v16bf b = load_b_pk(Bnode, ch * 4 + j, 4, ks, lane);
      acc[j] = wmma_bf(a, b, acc[j]);
    }
  }
  const int hi = lane >> 4, nl = lane & 15;
#pragma unroll
  for (int j = 0; j < 4; ++j) {
    int n = (ch * 4 + j) * 16 + nl;
    const float* wn = W_node + n * 132;   // f_nuc columns 0..3
#pragma unroll
    for (int v = 0; v < 8; ++v) {
      int m = rowBase + rt * 16 + hi * 8 + v;
      const float* fn = f_nuc + m * 4;
      float val = acc[j][v] + fn[0] * wn[0] + fn[1] * wn[1] + fn[2] * wn[2] + fn[3] * wn[3];
      nuc[m * HDIM + n] = f2bf(fmaxf(val, 0.0f));
    }
  }
}

// ---------------------------------------------------------------------------
// LSTM input projection: P[r, c] = nuc[r] @ Wih.T + bih[c] + bhh[c]  (f32 out)
// ---------------------------------------------------------------------------
__global__ __launch_bounds__(256) void k_lstm_pre(const u16* __restrict__ nuc,
                                                  const u16* __restrict__ Bl,
                                                  const float* __restrict__ bih,
                                                  const float* __restrict__ bhh,
                                                  float* __restrict__ P) {
  const int tid = threadIdx.x, lane = tid & 31, w = tid >> 5;
  const int rowBase = blockIdx.x * 128 + w * 16;
  const int ctBase = blockIdx.y * 2;
  const int row = rowBase + (lane & 15);
  const int khalf = (lane >> 4) * 8;
  v8f acc[2];
  acc[0] = vzero8(); acc[1] = vzero8();
#pragma unroll
  for (int ks = 0; ks < 4; ++ks) {
    v16bf a = load_a_glb(nuc, row, ks * 32 + khalf, HDIM);
#pragma unroll
    for (int j = 0; j < 2; ++j) {
      v16bf b = load_b_pk(Bl, ctBase + j, 4, ks, lane);
      acc[j] = wmma_bf(a, b, acc[j]);
    }
  }
  const int hi = lane >> 4, nl = lane & 15;
#pragma unroll
  for (int j = 0; j < 2; ++j) {
    int c = (ctBase + j) * 16 + nl;
    float bb = bih[c] + bhh[c];
#pragma unroll
    for (int v = 0; v < 8; ++v) {
      int m = rowBase + hi * 8 + v;
      P[m * 256 + c] = acc[j][v] + bb;
    }
  }
}

// ---------------------------------------------------------------------------
// Bidirectional LSTM recurrence + fused max-pool.
// One block per (batch, direction); 256 threads = 256 gate columns.
// ---------------------------------------------------------------------------
__global__ __launch_bounds__(256) void k_lstm_seq(const float* __restrict__ Pf,
                                                  const float* __restrict__ Pb,
                                                  const float* __restrict__ Whh_f,
                                                  const float* __restrict__ Whh_b,
                                                  float* __restrict__ out) {
  __shared__ float h_s[64];
  __shared__ float g_s[256];
  const int tid = threadIdx.x;
  const int b = blockIdx.x & 127;
  const int dir = blockIdx.x >> 7;
  const float* P   = dir ? Pb : Pf;
  const float* Whh = dir ? Whh_b : Whh_f;
  const float* wr  = Whh + tid * 64;

  float c_reg = 0.0f, maxv = -3.4e38f;
  if (tid < 64) h_s[tid] = 0.0f;
  __syncthreads();

  for (int t = 0; t < 512; ++t) {
    int tt = dir ? (511 - t) : t;
    float g = P[(b * 512 + tt) * 256 + tid];
#pragma unroll 8
    for (int k = 0; k < 64; ++k) g = fmaf(wr[k], h_s[k], g);
    g_s[tid] = g;
    __syncthreads();
    if (tid < 64) {
      float gi = g_s[tid], gf = g_s[64 + tid], gg = g_s[128 + tid], go = g_s[192 + tid];
      c_reg = sigf(gf) * c_reg + sigf(gi) * tanhf(gg);
      float hh = sigf(go) * tanhf(c_reg);
      maxv = fmaxf(maxv, hh);
      h_s[tid] = hh;
    }
    __syncthreads();
  }
  if (tid < 64) out[b * 128 + dir * 64 + tid] = maxv;
}

// ---------------------------------------------------------------------------
// Host-side orchestration
// ---------------------------------------------------------------------------
extern "C" void kernel_launch(void* const* d_in, const int* in_sizes, int n_in,
                              void* d_out, int out_size, void* d_ws, size_t ws_size,
                              hipStream_t stream) {
  const float* f_nuc   = (const float*)d_in[0];
  const float* f_bond  = (const float*)d_in[1];
  const int*   ngraph  = (const int*)d_in[2];
  const int*   mgraph  = (const int*)d_in[3];
  const float* W_local = (const float*)d_in[5];
  const float* W_msg   = (const float*)d_in[6];
  const float* W_node  = (const float*)d_in[7];
  const float* gWih    = (const float*)d_in[8];
  const float* gWhh    = (const float*)d_in[9];
  const float* gbih    = (const float*)d_in[10];
  const float* gbhh    = (const float*)d_in[11];
  const float* lWih_f  = (const float*)d_in[12];
  const float* lWhh_f  = (const float*)d_in[13];
  const float* lbih_f  = (const float*)d_in[14];
  const float* lbhh_f  = (const float*)d_in[15];
  const float* lWih_b  = (const float*)d_in[16];
  const float* lWhh_b  = (const float*)d_in[17];
  const float* lbih_b  = (const float*)d_in[18];
  const float* lbhh_b  = (const float*)d_in[19];
  float* out = (float*)d_out;

  // Workspace carve-up (~303 MB)
  char* base = (char*)d_ws;
  size_t off = 0;
  auto carve = [&](size_t bytes) -> void* {
    void* p = base + off;
    off = (off + bytes + 255) & ~(size_t)255;
    return p;
  };
  u16*   msg_bf   = (u16*)carve((size_t)M_MSG * HDIM * 2);
  u16*   local_bf = (u16*)carve((size_t)M_MSG * HDIM * 2);
  u16*   newmsg   = (u16*)carve((size_t)M_MSG * HDIM * 2);
  u16*   nuc_bf   = (u16*)carve((size_t)N_NUC * HDIM * 2);
  float* Pf       = (float*)carve((size_t)N_NUC * 256 * 4);
  float* Pb       = (float*)carve((size_t)N_NUC * 256 * 4);
  u16*   PKmsg    = (u16*)carve(8  * 4 * 512 * 2);
  u16*   PKgru    = (u16*)carve(32 * 8 * 512 * 2);
  u16*   PKnode   = (u16*)carve(8  * 4 * 512 * 2);
  u16*   PKlf     = (u16*)carve(16 * 4 * 512 * 2);
  u16*   PKlb     = (u16*)carve(16 * 4 * 512 * 2);
  (void)ws_size; (void)in_sizes; (void)n_in; (void)out_size;

  // --- Pack weights into WMMA B-fragment layout -----------------------------
  k_pack<<<64, 256, 0, stream>>>(PKmsg, 4, 0, W_msg, 128, 0, 4, 0, 8 * 4 * 512);
  // GRU combined: tiles 0..15 = r,z (Wih rows 0..255 at ks0..3, Whh at ks4..7)
  k_pack<<<128, 256, 0, stream>>>(PKgru, 8, 0, gWih, 128, 0, 4, 0, 16 * 4 * 512);
  k_pack<<<128, 256, 0, stream>>>(PKgru, 8, 4, gWhh, 128, 0, 4, 0, 16 * 4 * 512);
  // tiles 16..23 = i_n (Wih rows 256..383, ks0..3)
  k_pack<<<64, 256, 0, stream>>>(PKgru + 16 * 8 * 512, 8, 0, gWih, 128, 256, 4, 0, 8 * 4 * 512);
  // tiles 24..31 = h_n (Whh rows 256..383, ks4..7)
  k_pack<<<64, 256, 0, stream>>>(PKgru + 24 * 8 * 512, 8, 4, gWhh, 128, 256, 4, 0, 8 * 4 * 512);
  // W_node message-part (columns 4..131)
  k_pack<<<64, 256, 0, stream>>>(PKnode, 4, 0, W_node, 132, 0, 4, 4, 8 * 4 * 512);
  // LSTM input weights
  k_pack<<<128, 256, 0, stream>>>(PKlf, 4, 0, lWih_f, 128, 0, 4, 0, 16 * 4 * 512);
  k_pack<<<128, 256, 0, stream>>>(PKlb, 4, 0, lWih_b, 128, 0, 4, 0, 16 * 4 * 512);

  // --- local features + initial messages ------------------------------------
  k_local<<<(M_MSG * HDIM) / 256, 256, 0, stream>>>(f_bond, W_local, local_bf, msg_bf);

  // --- 5 message-passing iterations ------------------------------------------
  for (int it = 1; it < 6; ++it) {
    k_gemm_msg<<<M_MSG / 64, 256, 0, stream>>>(msg_bf, mgraph, PKmsg, local_bf, newmsg);
    k_gemm_gru<<<M_MSG / 32, 256, 0, stream>>>(newmsg, msg_bf, PKgru, gbih, gbhh);
  }

  // --- node embedding ---------------------------------------------------------
  k_gemm_node<<<N_NUC / 64, 256, 0, stream>>>(msg_bf, ngraph, PKnode, f_nuc, W_node, nuc_bf);

  // --- LSTM input projections (biases folded) ---------------------------------
  dim3 gpre(N_NUC / 128, 8);
  k_lstm_pre<<<gpre, 256, 0, stream>>>(nuc_bf, PKlf, lbih_f, lbhh_f, Pf);
  k_lstm_pre<<<gpre, 256, 0, stream>>>(nuc_bf, PKlb, lbih_b, lbhh_b, Pb);

  // --- recurrence + fused max-pool --------------------------------------------
  k_lstm_seq<<<256, 256, 0, stream>>>(Pf, Pb, lWhh_f, lWhh_b, out);
}
